// GaussianEmbedding_68762426409583
// MI455X (gfx1250) — matrix operand order repaired and checked by reading, not compiled, not run on hardware
//
#include <hip/hip_runtime.h>

// Problem constants (match reference)
#define BB    32
#define SS    512
#define S2V   514
#define KPAD  544          // 17 * 32, padded token count (K dimension)
#define AROW  552          // wA row stride in halves (1104B: bank-friendly, 16B-aligned)
#define BROW  40           // tileB row stride in halves (80B: bank-friendly, 16B-aligned)
#define DIM   256          // embedding dim (N)
#define TT    4096         // frames
#define TM    64           // frame rows per block
#define NCHUNK 17          // KPAD / 32
#define EPSF  1e-6f
#define INV_SQRT_2PI 0.3989422804014327f

typedef __attribute__((ext_vector_type(16))) _Float16 v16h;
typedef __attribute__((ext_vector_type(8)))  _Float16 v8h;
typedef __attribute__((ext_vector_type(8)))  float    v8f;

// ---------------------------------------------------------------------------
// Prep: per-batch exclusive cumsum of durations; per-token center, 1/sigma,
// amplitude (zeroed for pad tokens / padding slots); total duration per row.
// ---------------------------------------------------------------------------
__global__ void gauss_prep_kernel(const int* __restrict__ durs,
                                  float* __restrict__ wsC,
                                  float* __restrict__ wsInv,
                                  float* __restrict__ wsAmp,
                                  int*   __restrict__ wsTot) {
    __shared__ float excl[KPAD];
    const int b = blockIdx.x;
    if (threadIdx.x == 0) {
        float run = 0.f;                      // exact: integer sums <= 4096 in f32
        for (int s = 0; s < KPAD; ++s) {
            excl[s] = run;
            float d = (s < SS) ? (float)durs[b * SS + s] : 0.f;
            run += d;
        }
        wsTot[b] = (int)run;
    }
    __syncthreads();
    for (int s = (int)threadIdx.x; s < KPAD; s += (int)blockDim.x) {
        float d   = (s < SS) ? (float)durs[b * SS + s] : 0.f;
        float sig = d * 0.5f + EPSF;          // SIGMA_C = 2
        float inv = 1.0f / sig;
        wsC  [b * KPAD + s] = d * 0.5f + excl[s];
        wsInv[b * KPAD + s] = inv;
        // text ids for s < 512 are in [1, IDIM) (never PAD); s >= 512 are pad.
        wsAmp[b * KPAD + s] = (s < SS) ? inv * INV_SQRT_2PI : 0.f;
    }
}

// ---------------------------------------------------------------------------
// Main fused kernel: per (batch, 64-frame tile):
//   phase 1: unnormalized gaussian weights -> f16 LDS [64][AROW], row sums
//   phase 2: software-pipelined K-loop WMMA (prefetch next emb tile in regs)
//   phase 3: scale rows by 1/(sum+eps) (or 0 past total duration), store f32
// ---------------------------------------------------------------------------
__global__ void __launch_bounds__(256, 1)
gauss_embed_main_kernel(const int*   __restrict__ text,
                        const float* __restrict__ embed,
                        const float* __restrict__ wsC,
                        const float* __restrict__ wsInv,
                        const float* __restrict__ wsAmp,
                        const int*   __restrict__ wsTot,
                        float*       __restrict__ out) {
    extern __shared__ char smem_raw[];
    _Float16* wA   = (_Float16*)smem_raw;                              // [TM][AROW]
    _Float16* tB   = (_Float16*)(smem_raw + (size_t)TM * AROW * 2);    // [DIM][BROW]
    float*    sC   = (float*)(smem_raw + (size_t)TM * AROW * 2 + (size_t)DIM * BROW * 2);
    float*    sInv = sC   + KPAD;
    float*    sAmp = sInv + KPAD;
    int*      sIds = (int*)(sAmp + KPAD);                              // [KPAD]
    float*    sScale = (float*)(sIds + KPAD);                          // [TM]

    const int b    = blockIdx.y;
    const int t0   = blockIdx.x * TM;
    const int tid  = threadIdx.x;
    const int lane = tid & 31;
    const int wv   = tid >> 5;                // 0..7
    const int total = wsTot[b];

    // Stage per-token params + token ids into LDS.
    for (int s = tid; s < KPAD; s += 256) {
        sC[s]   = wsC  [b * KPAD + s];
        sInv[s] = wsInv[b * KPAD + s];
        sAmp[s] = wsAmp[b * KPAD + s];
        sIds[s] = (s < SS) ? text[b * SS + s] : 0;   // pad -> id 0 (zero row)
    }
    __syncthreads();

    // ---- Phase 1: weights + row sums (each wave owns 8 frame rows) ----
    for (int r = 0; r < 8; ++r) {
        const int   tr   = wv * 8 + r;                      // 0..63
        const float tval = (float)(t0 + tr) + 0.5f;
        float sum = 0.f;
        for (int s = lane; s < KPAD; s += 32) {
            float z = (tval - sC[s]) * sInv[s];
            float w = sAmp[s] * __expf(-0.5f * z * z);
            sum += w;
            wA[tr * AROW + s] = (_Float16)w;
        }
        sum += __shfl_xor(sum, 16);
        sum += __shfl_xor(sum, 8);
        sum += __shfl_xor(sum, 4);
        sum += __shfl_xor(sum, 2);
        sum += __shfl_xor(sum, 1);
        if (lane == 0) {
            // frames >= total duration collapse onto the zero pad-embedding row
            sScale[tr] = ((t0 + tr) < total) ? (1.0f / (sum + EPSF)) : 0.f;
        }
    }

    // ---- Phase 2: software-pipelined WMMA K-loop ----
    const int mt = wv & 3;                    // M tile: rows mt*16 .. mt*16+15
    const int nh = wv >> 2;                   // N half: cols nh*128 .. +127
    v8f acc[8];
    #pragma unroll
    for (int j = 0; j < 8; ++j) { v8f z = {}; acc[j] = z; }

    const int mrow = mt * 16 + (lane & 15);
    const int kb   = (lane >> 4) * 8;         // A: lane-half K sub-offset
    const int kseg = (lane >> 4) * 16;        // B: lane-half K sub-offset

    // Prologue: prefetch chunk 0 of the embedding tile into registers.
    float pre[32];
    {
        const int d = tid;
        #pragma unroll
        for (int kk = 0; kk < 32; ++kk)
            pre[kk] = embed[sIds[kk] * DIM + d];
    }

    for (int kc = 0; kc < NCHUNK; ++kc) {
        __syncthreads();                      // all reads of tB (prev chunk) done
        // Commit prefetched chunk kc: cvt f32->f16 into transposed tile tB[d][kk]
        {
            const int d = tid;
            #pragma unroll
            for (int kk = 0; kk < 32; ++kk)
                tB[d * BROW + kk] = (_Float16)pre[kk];
        }
        __syncthreads();

        // Prefetch chunk kc+1 (global-load latency hides behind WMMAs + barrier)
        if (kc + 1 < NCHUNK) {
            const int d = tid;
            #pragma unroll
            for (int kk = 0; kk < 32; ++kk)
                pre[kk] = embed[sIds[(kc + 1) * 32 + kk] * DIM + d];
        }

        // A fragment: 16x32 f16 (lanes 0-15: K 0-15, 16-31: K 16-31)
        const _Float16* ap = wA + mrow * AROW + kc * 32;
        v8h alo = *(const v8h*)(ap + 2 * kb);
        v8h ahi = *(const v8h*)(ap + 2 * kb + 8);
        v16h afrag = __builtin_shufflevector(alo, ahi,
                        0,1,2,3,4,5,6,7,8,9,10,11,12,13,14,15);

        // Load ALL 8 B fragments into distinct registers first so the
        // ds-load waits cascade instead of serializing each WMMA.
        v16h bf[8];
        #pragma unroll
        for (int j = 0; j < 8; ++j) {
            const int n = nh * 128 + j * 16 + (lane & 15);
            const _Float16* bp = tB + n * BROW + kseg;   // 16 contiguous K halves
            v8h blo = *(const v8h*)(bp);
            v8h bhi = *(const v8h*)(bp + 8);
            bf[j] = __builtin_shufflevector(blo, bhi,
                        0,1,2,3,4,5,6,7,8,9,10,11,12,13,14,15);
        }
        #pragma unroll
        for (int j = 0; j < 8; ++j) {
            acc[j] = __builtin_amdgcn_wmma_f32_16x16x32_f16(
                         /*neg_a=*/false, afrag, /*neg_b=*/false, bf[j],
                         /*c_mod=*/(short)0, acc[j],
                         /*reuse_a=*/false, /*reuse_b=*/false);
        }
    }

    // ---- Phase 3: normalize rows & store ----
    #pragma unroll
    for (int j = 0; j < 8; ++j) {
        const int ncol = nh * 128 + j * 16 + (lane & 15);
        #pragma unroll
        for (int r = 0; r < 8; ++r) {
            const int rowRel = mt * 16 + r + ((lane >> 4) ? 8 : 0);  // C/D layout
            const float scale = sScale[rowRel];
            out[((size_t)b * TT + (t0 + rowRel)) * DIM + ncol] = acc[j][r] * scale;
        }
    }
}

// ---------------------------------------------------------------------------
extern "C" void kernel_launch(void* const* d_in, const int* in_sizes, int n_in,
                              void* d_out, int out_size, void* d_ws, size_t ws_size,
                              hipStream_t stream) {
    const int*   text  = (const int*)d_in[0];
    const int*   durs  = (const int*)d_in[1];
    const float* embed = (const float*)d_in[2];
    float*       out   = (float*)d_out;

    char*  ws    = (char*)d_ws;
    float* wsC   = (float*)ws;
    float* wsInv = wsC   + (size_t)BB * KPAD;
    float* wsAmp = wsInv + (size_t)BB * KPAD;
    int*   wsTot = (int*)(wsAmp + (size_t)BB * KPAD);

    gauss_prep_kernel<<<dim3(BB), dim3(256), 0, stream>>>(durs, wsC, wsInv, wsAmp, wsTot);

    const size_t smem = (size_t)TM * AROW * 2     // wA f16
                      + (size_t)DIM * BROW * 2    // tB f16
                      + (size_t)KPAD * 4 * 3      // sC / sInv / sAmp
                      + (size_t)KPAD * 4          // sIds
                      + (size_t)TM * 4;           // sScale
    gauss_embed_main_kernel<<<dim3(TT / TM, BB), dim3(256), smem, stream>>>(
        text, embed, wsC, wsInv, wsAmp, wsTot, out);
}
